// TSA_46385646797313
// MI455X (gfx1250) — matrix operand order repaired and checked by reading, not compiled, not run on hardware
//
#include <hip/hip_runtime.h>
#include <hip/hip_bf16.h>

typedef __bf16 bf16_t;
typedef __attribute__((ext_vector_type(16))) __bf16  v16bf;
typedef __attribute__((ext_vector_type(8)))  float   v8f;
typedef __attribute__((ext_vector_type(4)))  unsigned int u32x4;
typedef __attribute__((ext_vector_type(4)))  int   i32x4;
typedef __attribute__((ext_vector_type(8)))  int   i32x8;

#define B_SZ    2
#define L_SEQ   2048
#define D_MODEL 1024
#define H_HEADS 16
#define DHEAD   64

// ---------------------------------------------------------------------------
// WMMA helpers (bf16 A/B, fp32 accumulate): D = A(16x32) x B(32x16) + C
// A fragment (ISA 16-bit A 16x32 table): lane = m (0..15) twice; low lanes
// hold K {0..7,16..23}, high lanes K {8..15,24..31}  -> two 16B chunks.
// B fragment (same K order as A): lane n holds column n, low lanes
// K {0..7,16..23}, high lanes K {8..15,24..31} -> two 16B chunks.
// ---------------------------------------------------------------------------
union FragU { v16bf v; u32x4 u[2]; };

__device__ __forceinline__ v16bf loadA_frag(const bf16_t* m, int ld) {
    const int lane = threadIdx.x & 31;
    const int r    = lane & 15;
    const int kb   = (lane >> 4) * 8;
    const bf16_t* p = m + (size_t)r * ld;
    FragU f;
    f.u[0] = *reinterpret_cast<const u32x4*>(p + kb);
    f.u[1] = *reinterpret_cast<const u32x4*>(p + kb + 16);
    return f.v;
}

__device__ __forceinline__ v16bf loadB_frag(const bf16_t* m, int ld) {
    const int lane = threadIdx.x & 31;
    const int n    = lane & 15;
    const int kb   = (lane >> 4) * 8;
    const bf16_t* p = m + (size_t)n * ld + kb;
    FragU f;
    f.u[0] = *reinterpret_cast<const u32x4*>(p);
    f.u[1] = *reinterpret_cast<const u32x4*>(p + 16);
    return f.v;
}

__device__ __forceinline__ v8f wmma_bf16(v16bf a, v16bf b, v8f c) {
    return __builtin_amdgcn_wmma_f32_16x16x32_bf16(
        /*neg_a=*/false, a, /*neg_b=*/false, b,
        /*c_mod=*/(short)0, c, /*reuse_a=*/false, /*reuse_b=*/false);
}

// ---------------------------------------------------------------------------
// Tensor Data Mover: DMA a 64x64 bf16 tile (rows of 64 elems, row stride 64)
// from global into LDS, padding +16B after every 128B row -> LDS row stride
// 144B (= 72 bf16), matching the bank-conflict-reduced fragment reads.
// D# layout per CDNA5 ISA. Wave-level op,
// tracked with TENSORcnt.
// ---------------------------------------------------------------------------
__device__ __forceinline__ void tdm_load_tile64x64(const bf16_t* gsrc,
                                                   bf16_t* lds_dst) {
    const unsigned long long ga = (unsigned long long)(uintptr_t)gsrc;
    u32x4 g0;
    g0.x = 1u;                                   // count=1 (valid user D#)
    g0.y = (unsigned int)(uintptr_t)lds_dst;     // lds_addr (bytes)
    g0.z = (unsigned int)ga;                     // global_addr[31:0]
    g0.w = ((unsigned int)(ga >> 32) & 0x01FFFFFFu) | (2u << 30); // [56:32]|type=2
    i32x8 g1;
    g1[0] = (int)((1u << 16)      // data_size = 1 -> 2 bytes
                | (1u << 20)      // pad_enable
                | (4u << 22)      // pad_interval: code 4 -> every 32 DWORDs
                | (3u << 25));    // pad_amount:   code 3 -> 4 DWORDs (16B)
    g1[1] = (int)(64u << 16);     // tensor_dim0 = 64 (bits 79:48, low part)
    g1[2] = (int)(2048u << 16);   // dim0 hi=0 | tensor_dim1 = 2048 (low 16)
    g1[3] = (int)(64u << 16);     // dim1 hi=0 | tile_dim0 = 64
    g1[4] = 64;                   // tile_dim1 = 64, tile_dim2 = 0
    g1[5] = 64;                   // tensor_dim0_stride = 64 (low 32)
    g1[6] = 0;                    // stride hi | tensor_dim1_stride lo
    g1[7] = 0;
    i32x4 z4 = {0, 0, 0, 0};
#if defined(__clang_major__) && (__clang_major__ >= 23)
    i32x8 z8 = {0, 0, 0, 0, 0, 0, 0, 0};
    __builtin_amdgcn_tensor_load_to_lds(g0, g1, z4, z4, z8, 0);
#else
    __builtin_amdgcn_tensor_load_to_lds(g0, g1, z4, z4, 0);
#endif
}

// ---------------------------------------------------------------------------
// fp32 -> bf16 conversion
// ---------------------------------------------------------------------------
__global__ void cvt_bf16_kernel(const float* __restrict__ in,
                                bf16_t* __restrict__ out, int n) {
    int i = blockIdx.x * blockDim.x + threadIdx.x;
    if (i < n) out[i] = (bf16_t)in[i];
}

// ---------------------------------------------------------------------------
// Saliency conv: Conv1d(D->H, k=3, pad=1) over src [B,L,D] -> sal [B,H,L]
// ---------------------------------------------------------------------------
__global__ void sal_kernel(const float* __restrict__ src,
                           const float* __restrict__ conv_w,
                           const float* __restrict__ conv_b,
                           float* __restrict__ sal) {
    const int idx = blockIdx.x * blockDim.x + threadIdx.x;  // B*H*L
    const int b = idx >> 15;            // H*L = 32768
    const int h = (idx >> 11) & 15;     // L   = 2048
    const int l = idx & (L_SEQ - 1);
    const float* r1 = src + ((size_t)b * L_SEQ + l) * D_MODEL;
    const float* r0 = (l > 0)         ? r1 - D_MODEL : r1;
    const float* r2 = (l < L_SEQ - 1) ? r1 + D_MODEL : r1;
    const float m0 = (l > 0) ? 1.f : 0.f;
    const float m2 = (l < L_SEQ - 1) ? 1.f : 0.f;
    const float* w = conv_w + (size_t)h * D_MODEL * 3;
    float s = conv_b[h];
    for (int d = 0; d < D_MODEL; ++d)
        s += m0 * r0[d] * w[d*3+0] + r1[d] * w[d*3+1] + m2 * r2[d] * w[d*3+2];
    sal[idx] = s;
}

// ---------------------------------------------------------------------------
// GEMM: C[M,1024] = A[M,1024] @ W[1024,1024]^T + bias,  bf16 in / fp32 acc.
// mode 0: store bf16 head-split  [B,H,L,DH]
// mode 1: store bf16 transposed  [B,H,DH,L]   (for V -> row-major B operand)
// mode 2: store fp32 row-major   [M,1024]     (final output)
// Block: 128 thr = 4 waves; block tile 64x64; wave tile 16x64.
// ---------------------------------------------------------------------------
__global__ __launch_bounds__(128) void gemm_bf16_wmma(
    const bf16_t* __restrict__ A, const bf16_t* __restrict__ W,
    const float* __restrict__ bias, void* __restrict__ outp, int mode) {
    const int w     = threadIdx.x >> 5;
    const int lane  = threadIdx.x & 31;
    const int mBase = blockIdx.y * 64 + 16 * w;
    const int nBase = blockIdx.x * 64;

    v8f acc[4] = {};
    for (int kc = 0; kc < D_MODEL / 32; ++kc) {
        v16bf a = loadA_frag(A + (size_t)mBase * D_MODEL + kc * 32, D_MODEL);
#pragma unroll
        for (int nt = 0; nt < 4; ++nt) {
            v16bf b = loadB_frag(W + (size_t)(nBase + nt * 16) * D_MODEL + kc * 32,
                                 D_MODEL);
            acc[nt] = wmma_bf16(a, b, acc[nt]);
        }
    }

    const int hi = lane >> 4, nl = lane & 15;
#pragma unroll
    for (int nt = 0; nt < 4; ++nt) {
        const int n = nBase + nt * 16 + nl;
        const float bv = bias[n];
#pragma unroll
        for (int j = 0; j < 8; ++j) {
            const int m = mBase + j + 8 * hi;
            const float v = acc[nt][j] + bv;
            if (mode == 2) {
                ((float*)outp)[(size_t)m * D_MODEL + n] = v;
            } else {
                const int b_ = m >> 11, l = m & (L_SEQ - 1);
                const int h = n >> 6, dh = n & (DHEAD - 1);
                const size_t idx = (mode == 0)
                    ? ((((size_t)b_ * H_HEADS + h) * L_SEQ + l) * DHEAD + dh)
                    : ((((size_t)b_ * H_HEADS + h) * DHEAD + dh) * L_SEQ + l);
                ((bf16_t*)outp)[idx] = (bf16_t)v;
            }
        }
    }
}

// ---------------------------------------------------------------------------
// Fused saliency-biased attention. One block = one (b,h) x 64-query tile.
// Two-pass flash softmax; TDM double-buffers the K tile in LDS; writes
// normalized attn (fp32) + bf16 context.
// ---------------------------------------------------------------------------
__device__ __forceinline__ void compute_S_tile(const bf16_t* ktile, int ldk,
                                               int w, const v16bf* qa,
                                               float* Sbuf) {
    const int lane = threadIdx.x & 31;
    const int hi = lane >> 4, nl = lane & 15;
#pragma unroll
    for (int nt = 0; nt < 4; ++nt) {
        v8f c = {};
#pragma unroll
        for (int ch = 0; ch < 2; ++ch) {
            v16bf b = loadB_frag(ktile + (size_t)(nt * 16) * ldk + ch * 32, ldk);
            c = wmma_bf16(qa[ch], b, c);
        }
#pragma unroll
        for (int j = 0; j < 8; ++j)
            Sbuf[(16 * w + j + 8 * hi) * 68 + nt * 16 + nl] = c[j];
    }
}

__global__ __launch_bounds__(128) void attn_kernel(
    const bf16_t* __restrict__ Q, const bf16_t* __restrict__ Km,
    const bf16_t* __restrict__ Vt, const float* __restrict__ sal,
    float* __restrict__ attnOut, bf16_t* __restrict__ ctx) {
    __shared__ float  Sbuf[64 * 68];        // 17408B, conflict-free c-store/scan
    __shared__ bf16_t Pbuf[64 * 72];        //  9216B, conflict-free A-frag reads
    __shared__ bf16_t Ktile[2][64 * 72];    // 18432B, TDM double buffer (pad 72)
    __shared__ float  salBuf[L_SEQ];        //  8192B
    __shared__ float  rowM[64], rowS[64], rowInv[64];

    const int tid  = threadIdx.x;
    const int w    = tid >> 5;
    const int lane = tid & 31;
    const int bh    = blockIdx.y;
    const int qBase = blockIdx.x * 64;
    const int NKT   = L_SEQ / 64;

    const bf16_t* Qbh  = Q  + (size_t)bh * L_SEQ * DHEAD;
    const bf16_t* Kbh  = Km + (size_t)bh * L_SEQ * DHEAD;
    const bf16_t* Vtbh = Vt + (size_t)bh * DHEAD * L_SEQ;
    const float*  salbh = sal + (size_t)bh * L_SEQ;

    // kick off DMA of K tile 0 ASAP; it overlaps the setup below
    if (w == 0) tdm_load_tile64x64(Kbh, Ktile[0]);

    for (int i = tid; i < L_SEQ; i += 128) salBuf[i] = salbh[i];
    if (tid < 64) { rowM[tid] = -1e30f; rowS[tid] = 0.f; }

    v16bf qa[2];
#pragma unroll
    for (int ch = 0; ch < 2; ++ch)
        qa[ch] = loadA_frag(Qbh + (size_t)(qBase + 16 * w) * DHEAD + ch * 32, DHEAD);

    if (w == 0) __builtin_amdgcn_s_wait_tensorcnt((short)0);
    __syncthreads();

    // ---- pass 1: per-row running max / sum-exp over all key tiles ----
    for (int kt = 0; kt < NKT; ++kt) {
        const int cur = kt & 1;
        if (w == 0 && kt + 1 < NKT)   // prefetch next K tile during compute
            tdm_load_tile64x64(Kbh + (size_t)(kt + 1) * 64 * DHEAD, Ktile[cur ^ 1]);
        compute_S_tile(Ktile[cur], 72, w, qa, Sbuf);
        __syncthreads();
        if (tid < 64) {
            const int r = tid;
            float* row = &Sbuf[r * 68];
            const float* sb = &salBuf[kt * 64];
            float mx = -1e30f;
#pragma unroll 4
            for (int j = 0; j < 64; ++j) {
                float v = row[j] * 0.125f + sb[j];   // 1/sqrt(64) + saliency bias
                row[j] = v;
                mx = fmaxf(mx, v);
            }
            const float nm = fmaxf(rowM[r], mx);
            float acc = rowS[r] * __expf(rowM[r] - nm);
#pragma unroll 4
            for (int j = 0; j < 64; ++j) acc += __expf(row[j] - nm);
            rowM[r] = nm; rowS[r] = acc;
        }
        if (w == 0) __builtin_amdgcn_s_wait_tensorcnt((short)0);
        __syncthreads();   // publish next tile + finish scan
    }
    if (tid < 64) rowInv[tid] = 1.0f / rowS[tid];
    if (w == 0) tdm_load_tile64x64(Kbh, Ktile[0]);   // restart for pass 2
    if (w == 0) __builtin_amdgcn_s_wait_tensorcnt((short)0);
    __syncthreads();

    // ---- pass 2: normalized probs -> HBM attn + P @ V^T accumulation ----
    v8f oacc[4] = {};
    for (int kt = 0; kt < NKT; ++kt) {
        const int cur = kt & 1;
        if (w == 0 && kt + 1 < NKT)
            tdm_load_tile64x64(Kbh + (size_t)(kt + 1) * 64 * DHEAD, Ktile[cur ^ 1]);
        compute_S_tile(Ktile[cur], 72, w, qa, Sbuf);
        __syncthreads();
        {
            const int r  = tid >> 1;
            const int c0 = (tid & 1) * 32;
            const float m = rowM[r], inv = rowInv[r];
            const float* row = &Sbuf[r * 68];
            const float* sb  = &salBuf[kt * 64];
            float* arow = attnOut + ((size_t)bh * L_SEQ + qBase + r) * L_SEQ + kt * 64;
#pragma unroll 4
            for (int j = c0; j < c0 + 32; ++j) {
                const float v = row[j] * 0.125f + sb[j];
                const float p = __expf(v - m) * inv;
                arow[j] = p;
                Pbuf[r * 72 + j] = (bf16_t)p;
            }
        }
        __syncthreads();
#pragma unroll
        for (int nt = 0; nt < 4; ++nt) {
#pragma unroll
            for (int ch = 0; ch < 2; ++ch) {
                v16bf a = loadA_frag(&Pbuf[16 * w * 72 + ch * 32], 72);
                v16bf b = loadB_frag(Vtbh + (size_t)(nt * 16) * L_SEQ + kt * 64 + ch * 32,
                                     L_SEQ);
                oacc[nt] = wmma_bf16(a, b, oacc[nt]);
            }
        }
        if (w == 0) __builtin_amdgcn_s_wait_tensorcnt((short)0);
        __syncthreads();   // publish next K tile; Pbuf reads done before rewrite
    }

    // ---- store bf16 context [B, L, D] (head-interleaved) ----
    const int hi = lane >> 4, nl = lane & 15;
    const int b_ = bh / H_HEADS, h = bh % H_HEADS;
#pragma unroll
    for (int nt = 0; nt < 4; ++nt) {
#pragma unroll
        for (int j = 0; j < 8; ++j) {
            const int q = qBase + 16 * w + j + 8 * hi;
            const int d = nt * 16 + nl;
            ctx[((size_t)b_ * L_SEQ + q) * D_MODEL + h * DHEAD + d] = (bf16_t)oacc[nt][j];
        }
    }
}

// ---------------------------------------------------------------------------
// Host launcher
// ---------------------------------------------------------------------------
extern "C" void kernel_launch(void* const* d_in, const int* in_sizes, int n_in,
                              void* d_out, int out_size, void* d_ws, size_t ws_size,
                              hipStream_t stream) {
    (void)in_sizes; (void)n_in; (void)out_size; (void)ws_size;
    const float* src    = (const float*)d_in[0];
    const float* Wq     = (const float*)d_in[1];
    const float* bq     = (const float*)d_in[2];
    const float* Wk     = (const float*)d_in[3];
    const float* bk     = (const float*)d_in[4];
    const float* Wv     = (const float*)d_in[5];
    const float* bv     = (const float*)d_in[6];
    const float* Wo     = (const float*)d_in[7];
    const float* bo     = (const float*)d_in[8];
    const float* conv_w = (const float*)d_in[9];
    const float* conv_b = (const float*)d_in[10];

    const size_t MB = (size_t)1 << 20;
    char* ws = (char*)d_ws;
    bf16_t* srcb = (bf16_t*)(ws);             // [4096,1024]       8 MiB
    bf16_t* wqb  = (bf16_t*)(ws +  8 * MB);   // [1024,1024]       2 MiB
    bf16_t* wkb  = (bf16_t*)(ws + 10 * MB);
    bf16_t* wvb  = (bf16_t*)(ws + 12 * MB);
    bf16_t* wob  = (bf16_t*)(ws + 14 * MB);
    bf16_t* Qb   = (bf16_t*)(ws + 16 * MB);   // [B,H,L,DH]        8 MiB
    bf16_t* Kb   = (bf16_t*)(ws + 24 * MB);   // [B,H,L,DH]        8 MiB
    bf16_t* Vtb  = (bf16_t*)(ws + 32 * MB);   // [B,H,DH,L]        8 MiB
    bf16_t* ctxb = (bf16_t*)(ws + 40 * MB);   // [B,L,D]           8 MiB
    float*  salw = (float*)(ws + 48 * MB);    // [B,H,L]         256 KiB

    float* outF    = (float*)d_out;                                   // [B,L,D]
    float* attnOut = outF + (size_t)B_SZ * L_SEQ * D_MODEL;           // [B,H,L,L]

    const int nSrc = B_SZ * L_SEQ * D_MODEL;      // 4,194,304
    const int nW   = D_MODEL * D_MODEL;           // 1,048,576

    cvt_bf16_kernel<<<nSrc / 256, 256, 0, stream>>>(src, srcb, nSrc);
    cvt_bf16_kernel<<<nW / 256, 256, 0, stream>>>(Wq, wqb, nW);
    cvt_bf16_kernel<<<nW / 256, 256, 0, stream>>>(Wk, wkb, nW);
    cvt_bf16_kernel<<<nW / 256, 256, 0, stream>>>(Wv, wvb, nW);
    cvt_bf16_kernel<<<nW / 256, 256, 0, stream>>>(Wo, wob, nW);

    sal_kernel<<<(B_SZ * H_HEADS * L_SEQ) / 256, 256, 0, stream>>>(
        src, conv_w, conv_b, salw);

    const dim3 gGemm(D_MODEL / 64, (B_SZ * L_SEQ) / 64);   // (16, 64)
    gemm_bf16_wmma<<<gGemm, 128, 0, stream>>>(srcb, wqb, bq, Qb, 0);
    gemm_bf16_wmma<<<gGemm, 128, 0, stream>>>(srcb, wkb, bk, Kb, 0);
    gemm_bf16_wmma<<<gGemm, 128, 0, stream>>>(srcb, wvb, bv, Vtb, 1);

    const dim3 gAttn(L_SEQ / 64, B_SZ * H_HEADS);          // (32, 32)
    attn_kernel<<<gAttn, 128, 0, stream>>>(Qb, Kb, Vtb, salw, attnOut, ctxb);

    gemm_bf16_wmma<<<gGemm, 128, 0, stream>>>(ctxb, wob, bo, outF, 2);
}
